// SelfAttentionBlock_40870908789129
// MI455X (gfx1250) — matrix operand order repaired and checked by reading, not compiled, not run on hardware
//
#include <hip/hip_runtime.h>
#include <math.h>

// ---------------------------------------------------------------------------
// SelfAttentionBlock for MI455X (gfx1250): GroupNorm -> QKV (bf16 WMMA GEMM)
// -> fused attention (QK^T, softmax, AV) + output projection + residual.
// All matrix math via v_wmma_f32_16x16x32_bf16. All global WMMA operands are
// contiguous 32B-per-lane (paired global_load_b128). Q row-block staged into
// LDS via the Tensor Data Mover when the builtin is available.
// ---------------------------------------------------------------------------

#define B_   64
#define C_   256
#define HW_  1024
#define G_   32

typedef __attribute__((ext_vector_type(16))) __bf16       v16bf;
typedef __attribute__((ext_vector_type(8)))  float        v8f;
typedef __attribute__((ext_vector_type(4)))  unsigned int v4u;
typedef __attribute__((ext_vector_type(8)))  int          v8i;
typedef __attribute__((ext_vector_type(4)))  int          v4i;

// --- WMMA operand loaders (contiguous k = 16*hi + e, same in A and B; wave32) -----------

// A-matrix 16x32 bf16 from row-major memory; src pre-offset to (row0, k0).
// Lane lo holds row M=lo; K set = 16*hi + {0..15} (two b128 per lane).
__device__ inline v16bf load_a_rm(const __bf16* src, int ld, int lane) {
  int lo = lane & 15, hi = lane >> 4;
  const __bf16* p = src + (long)lo * ld + hi * 16;
  v16bf a;
#pragma unroll
  for (int e = 0; e < 16; ++e) {
    a[e] = p[e];
  }
  return a;
}

// B-matrix 32x16 where memory is [N][K] row-major (B = M^T view):
// contiguous 16-element (32-byte) run per lane (two b128 per lane).
__device__ inline v16bf load_b_nm(const __bf16* src, long ld, int lane) {
  int lo = lane & 15, hi = lane >> 4;
  const __bf16* p = src + (long)lo * ld + hi * 16;
  v16bf b;
#pragma unroll
  for (int e = 0; e < 16; ++e) b[e] = p[e];
  return b;
}

__device__ inline v8f wmma_bf16(v16bf a, v16bf b, v8f c) {
  return __builtin_amdgcn_wmma_f32_16x16x32_bf16(false, a, false, b,
                                                 (short)0, c, false, false);
}

// ---------------------------------------------------------------------------
// Kernel 0: convert the four 256x256 fp32 weight matrices to bf16.
// dst layout: [wq | wk | wv | wo], 65536 elements each.
// ---------------------------------------------------------------------------
__global__ __launch_bounds__(256) void cvt_weights_kernel(
    const float* wq, const float* wk, const float* wv, const float* wo,
    __bf16* dst) {
  int i = blockIdx.x * 256 + threadIdx.x;   // 0 .. 262143
  int sel = i >> 16, j = i & 65535;
  const float* s = (sel == 0) ? wq : (sel == 1) ? wk : (sel == 2) ? wv : wo;
  dst[i] = (__bf16)s[j];
}

// ---------------------------------------------------------------------------
// Kernel 1: GroupNorm. One block per (batch, group): 8 channels x 1024 px.
// Emits xn fp32 (residual) and bf16 token-major xt (GEMM A operand).
// ---------------------------------------------------------------------------
__global__ __launch_bounds__(256) void groupnorm_kernel(
    const float* __restrict__ x, const float* __restrict__ gamma,
    const float* __restrict__ beta, float* __restrict__ xn,
    __bf16* __restrict__ xt) {
  int b = blockIdx.x >> 5;
  int g = blockIdx.x & 31;
  int tid = threadIdx.x;
  const float* xg = x + ((size_t)b * C_ + g * 8) * HW_;

  float v[32];
  float s = 0.f, s2 = 0.f;
#pragma unroll
  for (int i = 0; i < 32; ++i) {
    float val = xg[i * 256 + tid];
    v[i] = val;
    s += val;
    s2 += val * val;
  }
  __shared__ float r1[256], r2[256];
  r1[tid] = s; r2[tid] = s2;
  __syncthreads();
  for (int off = 128; off > 0; off >>= 1) {
    if (tid < off) { r1[tid] += r1[tid + off]; r2[tid] += r2[tid + off]; }
    __syncthreads();
  }
  float mean = r1[0] * (1.f / 8192.f);
  float var  = r2[0] * (1.f / 8192.f) - mean * mean;
  float rstd = rsqrtf(var + 1e-5f);

  float*  xno = xn + ((size_t)b * C_ + g * 8) * HW_;
  __bf16* xto = xt + (size_t)b * HW_ * C_;
#pragma unroll
  for (int i = 0; i < 32; ++i) {
    int idx = i * 256 + tid;
    int c = g * 8 + (idx >> 10);
    int n = idx & 1023;
    float y = (v[i] - mean) * rstd * gamma[c] + beta[c];
    xno[idx] = y;
    xto[(size_t)n * C_ + c] = (__bf16)y;
  }
}

// ---------------------------------------------------------------------------
// Kernel 2: QKV projections. One 16x16 WMMA tile per wave (K-loop 8 x k=32).
// All three share one form:  D[n][o] = sum_c xt[n][c] * W[o][c] + bias[o].
//   sel 0 -> Q  stored (HW, C)            (A operand of QK)
//   sel 1 -> K^T stored (HW, C)           (B operand of QK, [N][K] form)
//   sel 2 -> V^T stored (C, HW)           (B operand of AV, [N][K] form)
// V^T store goes through a per-wave LDS repack so global stores are b128.
// ---------------------------------------------------------------------------
__global__ __launch_bounds__(256) void qkv_kernel(
    const __bf16* __restrict__ xt,
    const __bf16* __restrict__ wqb, const __bf16* __restrict__ wkb,
    const __bf16* __restrict__ wvb, const float* __restrict__ bq,
    const float* __restrict__ bk, const float* __restrict__ bv,
    __bf16* __restrict__ qb, __bf16* __restrict__ ktb,
    __bf16* __restrict__ vtb) {
  __shared__ __bf16 sT[8][16 * 16];     // per-wave transpose scratch (V)
  int wid = threadIdx.x >> 5;
  int lane = threadIdx.x & 31;
  int gw = blockIdx.x * 8 + wid;        // 196608 global waves
  int b = gw / 3072;
  int t = gw % 3072;
  int sel = t >> 10;                    // 0=q 1=k 2=v
  int tt = t & 1023;
  int lo = lane & 15, hi = lane >> 4;

  int n0 = (tt >> 4) * 16;
  int o0 = (tt & 15) * 16;
  const __bf16* A = xt + ((size_t)b * HW_ + n0) * C_;
  const __bf16* W = (sel == 0) ? wqb : (sel == 1) ? wkb : wvb;
  v8f acc = {};
#pragma unroll
  for (int kk = 0; kk < 8; ++kk) {
    v16bf a  = load_a_rm(A + kk * 32, C_, lane);
    v16bf bm = load_b_nm(W + (long)o0 * C_ + kk * 32, C_, lane);
    acc = wmma_bf16(a, bm, acc);
  }
  float bias = ((sel == 0) ? bq : (sel == 1) ? bk : bv)[o0 + lo];

  if (sel != 2) {  // Q / K^T: direct (HW, C) store
    __bf16* D = ((sel == 0) ? qb : ktb) + ((size_t)b * HW_ + n0) * C_ + o0;
#pragma unroll
    for (int r = 0; r < 8; ++r)
      D[(long)(r + 8 * hi) * C_ + lo] = (__bf16)(acc[r] + bias);
  } else {         // V^T: repack through LDS, store b128 runs along tokens
    __bf16* st = sT[wid];
#pragma unroll
    for (int r = 0; r < 8; ++r)
      st[(r + 8 * hi) * 16 + lo] = (__bf16)(acc[r] + bias);
    unsigned p[4];
#pragma unroll
    for (int j = 0; j < 4; ++j) {
      union { __bf16 h[2]; unsigned u; } cv;
      cv.h[0] = st[(hi * 8 + 2 * j) * 16 + lo];
      cv.h[1] = st[(hi * 8 + 2 * j + 1) * 16 + lo];
      p[j] = cv.u;
    }
    v4u val = {p[0], p[1], p[2], p[3]};
    __bf16* D = vtb + (size_t)b * C_ * HW_ + (size_t)(o0 + lo) * HW_ + n0 +
                hi * 8;
    *(v4u*)D = val;
  }
}

// ---------------------------------------------------------------------------
// Kernel 3: fused attention + output projection + residual.
// One block (8 waves) per (batch, 16-token row block). Score row-block
// (16 x 1024) lives entirely in LDS; softmax normalization folded into the
// AV output scale so P stays unnormalized bf16. Q staged via TDM if present.
// ---------------------------------------------------------------------------
__global__ __launch_bounds__(256) void attn_kernel(
    const __bf16* __restrict__ qb, const __bf16* __restrict__ ktb,
    const __bf16* __restrict__ vtb, const __bf16* __restrict__ wob,
    const float* __restrict__ bo, const float* __restrict__ xn,
    float* __restrict__ out) {
  __shared__ __bf16 sQ[16 * 256];     //  8 KB  Q row block
  __shared__ float  sS[16 * 1024];    // 64 KB  raw scores
  __shared__ __bf16 sP[16 * 1024];    // 32 KB  exp(scores - max), unnormalized
  __shared__ __bf16 sOb[16 * 256];    //  8 KB  attention output (bf16)
  __shared__ float  red[16 * 16];
  __shared__ float  rowmax[16];
  __shared__ float  rowinv[16];

  int b  = blockIdx.x >> 6;
  int n0 = (blockIdx.x & 63) * 16;
  int tid = threadIdx.x, wid = tid >> 5, lane = tid & 31;
  int lo = lane & 15, hi = lane >> 4;

  // ---- stage Q row block (contiguous 8 KB) via the Tensor Data Mover -----
  const __bf16* Qg = qb + ((size_t)b * HW_ + n0) * C_;
#if defined(__gfx1250__) && __has_builtin(__builtin_amdgcn_tensor_load_to_lds)
  if (wid == 0) {
    unsigned lds = (unsigned)(size_t)(void*)sQ;
    unsigned long long ga = (unsigned long long)(size_t)Qg;
    // D# group 0: count=1 | lds_addr | global_addr[56:0] | type=2
    v4u g0 = {1u, lds, (unsigned)(ga & 0xffffffffu),
              (unsigned)((ga >> 32) & 0x1ffffffu) | (2u << 30)};
    // D# group 1: data_size=2B; 1-D tensor/tile of 4096 elements.
    v8i g1 = {(int)(1u << 16),        // [17:16] data_size = 1 (2 bytes)
              (int)(4096u << 16),     // tensor_dim0[15:0] (bits 63:48)
              (int)(1u << 16),        // tensor_dim1 = 1   (bits 111:80)
              (int)(4096u << 16),     // tile_dim0 = 4096  (bits 127:112)
              1,                      // tile_dim1 = 1     (bits 143:128)
              4096,                   // tensor_dim0_stride lo32
              0, 0};
    v4i gz = {0, 0, 0, 0};
#if __has_include(<hip/amd_detail/amd_gfx1250_TDM.h>)
    v8i gz8 = {0, 0, 0, 0, 0, 0, 0, 0};
    __builtin_amdgcn_tensor_load_to_lds(g0, g1, gz, gz, gz8, 0);
#else
    __builtin_amdgcn_tensor_load_to_lds(g0, g1, gz, gz, 0);
#endif
    __builtin_amdgcn_s_wait_tensorcnt(0);
  }
#else
  for (int i = tid; i < 16 * 256 / 4; i += 256)
    ((uint2*)sQ)[i] = ((const uint2*)Qg)[i];
#endif
  __syncthreads();

  // ---- phase 1: S = Q K^T  (each wave owns 128 score columns) ------------
  const __bf16* Kt = ktb + (size_t)b * HW_ * C_;   // (HW, C) = [N][K] form
  for (int ct = 0; ct < 8; ++ct) {
    int m0 = wid * 128 + ct * 16;
    v8f acc = {};
#pragma unroll
    for (int kk = 0; kk < 8; ++kk) {
      v16bf a  = load_a_rm(sQ + kk * 32, C_, lane);
      v16bf bm = load_b_nm(Kt + (long)m0 * C_ + kk * 32, C_, lane);
      acc = wmma_bf16(a, bm, acc);
    }
#pragma unroll
    for (int r = 0; r < 8; ++r)
      sS[(r + 8 * hi) * 1024 + m0 + lo] = acc[r];
  }
  __syncthreads();

  // ---- phase 2: row softmax stats; P = exp(s*scale - max) as bf16 --------
  {
    int row = tid >> 4, sub = tid & 15;
    const float scale = 0.125f;         // 1/sqrt(HID)
    const float log2e = 1.44269504f;
    float* srow = sS + row * 1024;
    float mx = -INFINITY;
    for (int j = 0; j < 64; ++j) mx = fmaxf(mx, srow[sub * 64 + j] * scale);
    red[row * 16 + sub] = mx;
    __syncthreads();
    if (sub == 0) {
      float m = red[row * 16];
      for (int j = 1; j < 16; ++j) m = fmaxf(m, red[row * 16 + j]);
      rowmax[row] = m;
    }
    __syncthreads();
    float rm = rowmax[row];
    float sum = 0.f;
    for (int j = 0; j < 64; ++j) {
      float sv = srow[sub * 64 + j] * scale;
      float p = exp2f((sv - rm) * log2e);
      sP[row * 1024 + sub * 64 + j] = (__bf16)p;
      sum += p;
    }
    red[row * 16 + sub] = sum;
    __syncthreads();
    if (sub == 0) {
      float tsum = 0.f;
      for (int j = 0; j < 16; ++j) tsum += red[row * 16 + j];
      rowinv[row] = 1.f / tsum;
    }
    __syncthreads();
  }

  // ---- phase 3: O = P V   (each wave owns 32 output channels) ------------
  const __bf16* Vt = vtb + (size_t)b * C_ * HW_;   // (C, HW) = [N][K] form
  {
    int o0 = wid * 32;
    v8f acc0 = {}, acc1 = {};
    for (int mt = 0; mt < 32; ++mt) {
      v16bf a  = load_a_rm(sP + mt * 32, 1024, lane);
      v16bf b0 = load_b_nm(Vt + (long)o0 * HW_ + mt * 32, HW_, lane);
      v16bf b1 = load_b_nm(Vt + (long)(o0 + 16) * HW_ + mt * 32, HW_, lane);
      acc0 = wmma_bf16(a, b0, acc0);
      acc1 = wmma_bf16(a, b1, acc1);
    }
#pragma unroll
    for (int r = 0; r < 8; ++r) {
      float inv = rowinv[r + 8 * hi];
      sOb[(r + 8 * hi) * 256 + o0 + lo]      = (__bf16)(acc0[r] * inv);
      sOb[(r + 8 * hi) * 256 + o0 + 16 + lo] = (__bf16)(acc1[r] * inv);
    }
  }
  __syncthreads();

  // ---- phase 4: Y = O Wo^T + bo + xn  (residual), scatter to (C, HW) -----
  {
    int d0 = wid * 32;
    v8f y0 = {}, y1 = {};
#pragma unroll
    for (int kk = 0; kk < 8; ++kk) {
      v16bf a  = load_a_rm(sOb + kk * 32, 256, lane);
      v16bf b0 = load_b_nm(wob + (long)d0 * 256 + kk * 32, 256, lane);
      v16bf b1 = load_b_nm(wob + (long)(d0 + 16) * 256 + kk * 32, 256, lane);
      y0 = wmma_bf16(a, b0, y0);
      y1 = wmma_bf16(a, b1, y1);
    }
    const float* Xr = xn + (size_t)b * C_ * HW_;
    float* Og = out + (size_t)b * C_ * HW_;
    float c0 = bo[d0 + lo], c1 = bo[d0 + 16 + lo];
#pragma unroll
    for (int r = 0; r < 8; ++r) {
      int n = n0 + r + 8 * hi;
      size_t i0 = (size_t)(d0 + lo) * HW_ + n;
      size_t i1 = (size_t)(d0 + 16 + lo) * HW_ + n;
      Og[i0] = y0[r] + c0 + Xr[i0];
      Og[i1] = y1[r] + c1 + Xr[i1];
    }
  }
}

// ---------------------------------------------------------------------------
// Host launcher
// ---------------------------------------------------------------------------
extern "C" void kernel_launch(void* const* d_in, const int* in_sizes, int n_in,
                              void* d_out, int out_size, void* d_ws,
                              size_t ws_size, hipStream_t stream) {
  const float* x     = (const float*)d_in[0];
  const float* wq    = (const float*)d_in[1];
  const float* bq    = (const float*)d_in[2];
  const float* wk    = (const float*)d_in[3];
  const float* bk    = (const float*)d_in[4];
  const float* wv    = (const float*)d_in[5];
  const float* bv    = (const float*)d_in[6];
  const float* wo    = (const float*)d_in[7];
  const float* bo    = (const float*)d_in[8];
  const float* gamma = (const float*)d_in[9];
  const float* beta  = (const float*)d_in[10];
  float* outp = (float*)d_out;

  char* ws = (char*)d_ws;
  float*  xn_f = (float*)ws;                        //  64 MB fp32 xn
  __bf16* xt_b = (__bf16*)(ws + 67108864);          //  32 MB (HW, C) bf16 xn
  __bf16* q_b  = (__bf16*)(ws + 100663296);         //  32 MB Q   (HW, C)
  __bf16* kt_b = (__bf16*)(ws + 134217728);         //  32 MB K^T (HW, C)
  __bf16* vt_b = (__bf16*)(ws + 167772160);         //  32 MB V^T (C, HW)
  __bf16* w_b  = (__bf16*)(ws + 201326592);         // 512 KB: wq|wk|wv|wo
  __bf16* wq_b = w_b;
  __bf16* wk_b = w_b + 65536;
  __bf16* wv_b = w_b + 131072;
  __bf16* wo_b = w_b + 196608;
  (void)in_sizes; (void)n_in; (void)out_size; (void)ws_size;

  cvt_weights_kernel<<<1024, 256, 0, stream>>>(wq, wk, wv, wo, w_b);
  groupnorm_kernel<<<B_ * G_, 256, 0, stream>>>(x, gamma, beta, xn_f, xt_b);
  qkv_kernel<<<24576, 256, 0, stream>>>(xt_b, wq_b, wk_b, wv_b, bq, bk, bv,
                                        q_b, kt_b, vt_b);
  attn_kernel<<<B_ * 64, 256, 0, stream>>>(q_b, kt_b, vt_b, wo_b, bo, xn_f,
                                           outp);
}
